// MHA_Block_171798692296
// MI455X (gfx1250) — matrix operand order repaired and checked by reading, not compiled, not run on hardware
//
#include <hip/hip_runtime.h>
#include <hip/hip_bf16.h>
#include <stdint.h>

typedef __bf16  bf16_t;
typedef __attribute__((ext_vector_type(16))) __bf16 v16bf;
typedef __attribute__((ext_vector_type(8)))  __bf16 v8bf;
typedef __attribute__((ext_vector_type(8)))  float  v8f;

#define BB   2
#define SS   2048
#define DIMM 1024
#define HH   16
#define QD   64
#define VD   96
#define HV   (HH * VD)   // 1536
#define SOFTCAP 5.0f

// ---------------------------------------------------------------------------
// WMMA helpers (CDNA5 wave32 layouts, CDNA5 ISA §7.12.2)
// ---------------------------------------------------------------------------
__device__ __forceinline__ v16bf load_a16x32(const bf16_t* rowk0, int hi) {
    // A 16x32 bf16: lane (m = lane&15). contiguous k = 16*hi .. 16*hi+15
    v8bf lo = *(const v8bf*)(rowk0 + 16 * hi);
    v8bf hh = *(const v8bf*)(rowk0 + 16 * hi + 8);
    v16bf r;
#pragma unroll
    for (int i = 0; i < 8; ++i) { r[i] = lo[i]; r[8 + i] = hh[i]; }
    return r;
}

__device__ __forceinline__ v8f wmma_bf16(v16bf a, v16bf b, v8f c) {
    return __builtin_amdgcn_wmma_f32_16x16x32_bf16(false, a, false, b, (short)0, c,
                                                   false, false);
}

// ---------------------------------------------------------------------------
// 1) RMSNorm: x[4096][1024] f32 -> xn bf16
// ---------------------------------------------------------------------------
__global__ void rmsnorm_kernel(const float* __restrict__ x,
                               const float* __restrict__ w,
                               bf16_t* __restrict__ xn) {
    int row = blockIdx.x;                 // 0..4095
    int t   = threadIdx.x;                // 0..255
    const float4* x4 = (const float4*)(x + (size_t)row * DIMM);
    const float4* w4 = (const float4*)w;
    float4 v = x4[t];
    float4 g = w4[t];
    float s = v.x * v.x + v.y * v.y + v.z * v.z + v.w * v.w;
    __shared__ float red[256];
    red[t] = s;
    __syncthreads();
#pragma unroll
    for (int off = 128; off > 0; off >>= 1) {
        if (t < off) red[t] += red[t + off];
        __syncthreads();
    }
    float scale = rsqrtf(red[0] * (1.0f / DIMM) + 1e-8f);
    bf16_t* o = xn + (size_t)row * DIMM + t * 4;
    o[0] = (bf16_t)(v.x * g.x * scale);
    o[1] = (bf16_t)(v.y * g.y * scale);
    o[2] = (bf16_t)(v.z * g.z * scale);
    o[3] = (bf16_t)(v.w * g.w * scale);
}

// ---------------------------------------------------------------------------
// 2) Weight convert + transpose: W[K][N] f32 -> WT[N][K] bf16
// ---------------------------------------------------------------------------
__global__ void convtrans_kernel(const float* __restrict__ W,
                                 bf16_t* __restrict__ WT, int K, int N) {
    size_t idx = (size_t)blockIdx.x * 256 + threadIdx.x;
    if (idx >= (size_t)K * N) return;
    int n = (int)(idx / K);
    int k = (int)(idx % K);
    WT[idx] = (bf16_t)W[(size_t)k * N + n];
}

// ---------------------------------------------------------------------------
// 3) bf16 WMMA GEMM, NT 16x16 tiles per wave (A-fragment reuse):
//    out[m][n] = sum_k A[m][k] * BT[n][k]
//    MODE 0: bf16 out [M][N]
//    MODE 1: V path: bf16 transposed out vT[b][n][s] + biasvec[n]
//    MODE 2: f32 out [M][N] + biasvec[n]
// ---------------------------------------------------------------------------
template <int NT, int MODE>
__global__ void gemm_bf16_kernel(const bf16_t* __restrict__ A,
                                 const bf16_t* __restrict__ BT,
                                 void* __restrict__ outp,
                                 const float* __restrict__ biasvec,
                                 int M, int N, int K) {
    int lane = threadIdx.x & 31;
    int wave = threadIdx.x >> 5;
    int grpn = (N >> 4) / NT;
    int tid = blockIdx.x * 8 + wave;
    int total = (M >> 4) * grpn;
    if (tid >= total) return;                 // wave-uniform exit
    int tg = tid % grpn, tm = tid / grpn;
    int n = lane & 15, hi = lane >> 4;

    const bf16_t* arow = A + (size_t)(tm * 16 + n) * K;
    const bf16_t* brow[NT];
#pragma unroll
    for (int t = 0; t < NT; ++t)
        brow[t] = BT + (size_t)((tg * NT + t) * 16 + n) * K + 16 * hi;

    v8f zero = {0.f, 0.f, 0.f, 0.f, 0.f, 0.f, 0.f, 0.f};
    v8f acc[NT];
#pragma unroll
    for (int t = 0; t < NT; ++t) acc[t] = zero;

    for (int k0 = 0; k0 < K; k0 += 32) {
        v16bf a = load_a16x32(arow + k0, hi);
#pragma unroll
        for (int t = 0; t < NT; ++t) {
            v16bf b = *(const v16bf*)(brow[t] + k0);
            acc[t] = wmma_bf16(a, b, acc[t]);
        }
    }

#pragma unroll
    for (int t = 0; t < NT; ++t) {
        int ng = (tg * NT + t) * 16 + n;
        if (MODE == 0) {
            bf16_t* out = (bf16_t*)outp;
#pragma unroll
            for (int r = 0; r < 8; ++r)
                out[(size_t)(tm * 16 + r + 8 * hi) * N + ng] = (bf16_t)acc[t][r];
        } else if (MODE == 1) {
            bf16_t* out = (bf16_t*)outp;
            float bvn = biasvec[ng];
#pragma unroll
            for (int r = 0; r < 8; ++r) {
                int mg = tm * 16 + r + 8 * hi;
                int b  = mg >> 11;                 // / S
                int s  = mg & (SS - 1);
                out[((size_t)b * VD + ng) * SS + s] = (bf16_t)(acc[t][r] + bvn);
            }
        } else {
            float* out = (float*)outp;
            float bn = biasvec[ng];
#pragma unroll
            for (int r = 0; r < 8; ++r)
                out[(size_t)(tm * 16 + r + 8 * hi) * N + ng] = acc[t][r] + bn;
        }
    }
}

// ---------------------------------------------------------------------------
// 4) RoPE. freq[d] = 2048^(-d/31) = 2^(-11 d / 31); deinterleave-style rope.
// ---------------------------------------------------------------------------
__global__ void rope_q_kernel(const bf16_t* __restrict__ qraw,
                              bf16_t* __restrict__ qrope) {
    int idx = blockIdx.x * 256 + threadIdx.x;     // B*S*H*32 = 2^21
    int d = idx & 31;
    int h = (idx >> 5) & 15;
    int s = (idx >> 9) & (SS - 1);
    int b = idx >> 20;
    const bf16_t* src = qraw + ((size_t)(b * SS + s)) * (HH * QD) + h * QD;
    float x1 = (float)src[2 * d], x2 = (float)src[2 * d + 1];
    float freq = exp2f((float)d * (-11.0f / 31.0f));
    float sn, cs;
    __sincosf((float)s * freq, &sn, &cs);
    bf16_t* dst = qrope + ((size_t)((b * HH + h) * SS + s)) * QD;
    dst[d]      = (bf16_t)(x1 * cs - x2 * sn);
    dst[d + 32] = (bf16_t)(x1 * sn + x2 * cs);
}

__global__ void rope_k_kernel(const bf16_t* __restrict__ kraw,
                              bf16_t* __restrict__ krope) {
    int idx = blockIdx.x * 256 + threadIdx.x;     // B*S*32 = 131072
    int d = idx & 31;
    int s = (idx >> 5) & (SS - 1);
    int b = idx >> 16;
    const bf16_t* src = kraw + ((size_t)(b * SS + s)) * QD;
    float x1 = (float)src[2 * d], x2 = (float)src[2 * d + 1];
    float freq = exp2f((float)d * (-11.0f / 31.0f));
    float sn, cs;
    __sincosf((float)s * freq, &sn, &cs);
    bf16_t* dst = krope + ((size_t)(b * SS + s)) * QD;
    dst[d]      = (bf16_t)(x1 * cs - x2 * sn);
    dst[d + 32] = (bf16_t)(x1 * sn + x2 * cs);
}

// ---------------------------------------------------------------------------
// 5) Attention. One wave = one (b,h, 16-query) tile, streaming over keys in
//    steps of 32. Softcap bounds logits to [-5,5] -> no running max needed.
//    The 536MB bias stream is double-buffered into LDS with
//    global_load_async_to_lds_b128 + s_wait_asynccnt (ASYNCcnt path), then
//    read as the WMMA C operand. Prefetch address wraps on the last
//    iteration so the loop body is branch-free.
// ---------------------------------------------------------------------------
__device__ __forceinline__ void async_bias_tile(const float* grow0,  // &bias[i0][j0]
                                                unsigned lds_base, int lane) {
    int r4 = lane >> 3;          // 0..3 : row within group of 4
    int c  = lane & 7;           // 0..7 : 16B column chunk
#pragma unroll
    for (int i = 0; i < 4; ++i) {
        int row = i * 4 + r4;
        unsigned lds = lds_base + (unsigned)(row * 128 + c * 16);
        uint64_t ga  = (uint64_t)(uintptr_t)(grow0 + (size_t)row * SS + c * 4);
        asm volatile("global_load_async_to_lds_b128 %0, %1, off"
                     :: "v"(lds), "v"(ga) : "memory");
    }
}

__global__ void attn_kernel(const bf16_t* __restrict__ qr,
                            const bf16_t* __restrict__ kr,
                            const bf16_t* __restrict__ vT,
                            const float* __restrict__ bias,
                            bf16_t* __restrict__ ctx) {
    int lane = threadIdx.x & 31;
    int wave = threadIdx.x >> 5;
    int id = blockIdx.x * 8 + wave;       // 0..4095 : (b,h,qt)
    int qt = id & 127;
    int h  = (id >> 7) & 15;
    int b  = id >> 11;
    int i0 = qt * 16;
    int n = lane & 15, hi = lane >> 4;

    __shared__ __align__(16) float  BiasLDS[8][2][16][32];  // 32 KB: dbl-buffered
    __shared__ __align__(64) bf16_t Pbuf[8][16][32];        //  8 KB: P staging
    bf16_t (*P)[32] = Pbuf[wave];
    unsigned lds_b[2] = { (unsigned)(uintptr_t)&BiasLDS[wave][0][0][0],
                          (unsigned)(uintptr_t)&BiasLDS[wave][1][0][0] };

    // Q fragments (hoisted): A 16x64 = two 16x32 frags
    const bf16_t* qbase = qr + ((size_t)(b * HH + h) * SS + i0 + n) * QD;
    v16bf aq0 = load_a16x32(qbase + 0,  hi);
    v16bf aq1 = load_a16x32(qbase + 32, hi);

    const float*  biasbase = bias + ((size_t)(b * HH + h) * SS + i0) * SS;
    const bf16_t* kbase = kr + (size_t)b * SS * QD;
    const bf16_t* vbase = vT + (size_t)b * VD * SS;

    v8f zero = {0.f, 0.f, 0.f, 0.f, 0.f, 0.f, 0.f, 0.f};
    v8f cacc[6] = {zero, zero, zero, zero, zero, zero};
    float den[8] = {0.f, 0.f, 0.f, 0.f, 0.f, 0.f, 0.f, 0.f};

    // preload first bias tile
    async_bias_tile(biasbase, lds_b[0], lane);

    int buf = 0;
    for (int j0 = 0; j0 < SS; j0 += 32) {
        // branch-free prefetch of next tile (wraps to tile 0 on last iter;
        // that buffer is never read and s_endpgm drains ASYNCcnt)
        async_bias_tile(biasbase + ((j0 + 32) & (SS - 1)), lds_b[buf ^ 1], lane);
        asm volatile("s_wait_asynccnt 0x4" ::: "memory");   // current tile landed
        const float (*bl)[32] = BiasLDS[wave][buf];

#pragma unroll
        for (int jt = 0; jt < 32; jt += 16) {
            v8f acc;
#pragma unroll
            for (int r = 0; r < 8; ++r)    // bias tile (from LDS) -> C operand
                acc[r] = bl[r + 8 * hi][jt + n];
            const bf16_t* krow = kbase + (size_t)(j0 + jt + n) * QD + 16 * hi;
            v16bf kb0 = *(const v16bf*)(krow);
            v16bf kb1 = *(const v16bf*)(krow + 32);
            acc = wmma_bf16(aq0, kb0, acc);
            acc = wmma_bf16(aq1, kb1, acc);
#pragma unroll
            for (int r = 0; r < 8; ++r) {
                float l = tanhf(acc[r] * (1.0f / SOFTCAP)) * SOFTCAP;
                float p = __expf(l);       // bounded by e^5, no max needed
                den[r] += p;
                P[r + 8 * hi][jt + n] = (bf16_t)p;
            }
        }
        __builtin_amdgcn_wave_barrier();   // LDS in-order per wave; block reorder
        v16bf ap = load_a16x32(&P[n][0], hi);
        // preload all 6 V B-fragments so loads overlap, then WMMA chain
        v16bf vb[6];
#pragma unroll
        for (int nt = 0; nt < 6; ++nt)
            vb[nt] = *(const v16bf*)(vbase + (size_t)(nt * 16 + n) * SS + j0 + 16 * hi);
#pragma unroll
        for (int nt = 0; nt < 6; ++nt)
            cacc[nt] = wmma_bf16(ap, vb[nt], cacc[nt]);
        __builtin_amdgcn_wave_barrier();
        buf ^= 1;
    }

    // softmax denominator: reduce across the 16 lanes of each half
#pragma unroll
    for (int r = 0; r < 8; ++r) {
        float d = den[r];
        d += __shfl_xor(d, 1, 32);
        d += __shfl_xor(d, 2, 32);
        d += __shfl_xor(d, 4, 32);
        d += __shfl_xor(d, 8, 32);
        den[r] = 1.0f / d;
    }

    bf16_t* cbase = ctx + ((size_t)(b * SS) + i0) * HV + h * VD;
#pragma unroll
    for (int nt = 0; nt < 6; ++nt)
#pragma unroll
        for (int r = 0; r < 8; ++r)
            cbase[(size_t)(r + 8 * hi) * HV + nt * 16 + n] =
                (bf16_t)(cacc[nt][r] * den[r]);
}

// ---------------------------------------------------------------------------
// Launch
// ---------------------------------------------------------------------------
extern "C" void kernel_launch(void* const* d_in, const int* in_sizes, int n_in,
                              void* d_out, int out_size, void* d_ws, size_t ws_size,
                              hipStream_t stream) {
    const float* x    = (const float*)d_in[0];
    const float* abias= (const float*)d_in[1];
    const float* rmsw = (const float*)d_in[2];
    const float* Wq   = (const float*)d_in[3];
    const float* Wk   = (const float*)d_in[4];
    const float* Wv   = (const float*)d_in[5];
    const float* bv   = (const float*)d_in[6];
    const float* Wo   = (const float*)d_in[7];
    const float* bo   = (const float*)d_in[8];
    float* out = (float*)d_out;

    char* ws = (char*)d_ws;
    size_t off = 0;
    auto carve = [&](size_t bytes) -> void* {
        void* p = ws + off;
        off += (bytes + 255) & ~(size_t)255;
        return p;
    };
    const size_t M = (size_t)BB * SS;                  // 4096
    bf16_t* xn    = (bf16_t*)carve(M * DIMM * 2);
    bf16_t* WqT   = (bf16_t*)carve((size_t)DIMM * DIMM * 2);
    bf16_t* WkT   = (bf16_t*)carve((size_t)QD * DIMM * 2);
    bf16_t* WvT   = (bf16_t*)carve((size_t)VD * DIMM * 2);
    bf16_t* WoT   = (bf16_t*)carve((size_t)DIMM * HV * 2);
    bf16_t* qraw  = (bf16_t*)carve(M * DIMM * 2);
    bf16_t* kraw  = (bf16_t*)carve(M * QD * 2);
    bf16_t* vTbuf = (bf16_t*)carve((size_t)BB * VD * SS * 2);
    bf16_t* qrope = (bf16_t*)carve((size_t)BB * HH * SS * QD * 2);
    bf16_t* krope = (bf16_t*)carve((size_t)BB * SS * QD * 2);
    bf16_t* ctx   = (bf16_t*)carve(M * HV * 2);
    (void)in_sizes; (void)n_in; (void)out_size; (void)ws_size;

    auto cdiv = [](size_t a, size_t b) { return (int)((a + b - 1) / b); };

    // 1) RMSNorm
    rmsnorm_kernel<<<(int)M, 256, 0, stream>>>(x, rmsw, xn);

    // 2) weight transposes
    convtrans_kernel<<<cdiv((size_t)DIMM * DIMM, 256), 256, 0, stream>>>(Wq, WqT, DIMM, DIMM);
    convtrans_kernel<<<cdiv((size_t)DIMM * QD,   256), 256, 0, stream>>>(Wk, WkT, DIMM, QD);
    convtrans_kernel<<<cdiv((size_t)DIMM * VD,   256), 256, 0, stream>>>(Wv, WvT, DIMM, VD);
    convtrans_kernel<<<cdiv((size_t)HV * DIMM,   256), 256, 0, stream>>>(Wo, WoT, HV, DIMM);

    // 3) QKV projections (NT tiles per wave)
    {
        int tiles = (int)(M / 16) * ((DIMM / 16) / 4);   // q: 16x64 per wave
        gemm_bf16_kernel<4, 0><<<cdiv(tiles, 8), 256, 0, stream>>>(xn, WqT, qraw, nullptr,
                                                                   (int)M, DIMM, DIMM);
    }
    {
        int tiles = (int)(M / 16) * ((QD / 16) / 4);     // k: one 16x64 group
        gemm_bf16_kernel<4, 0><<<cdiv(tiles, 8), 256, 0, stream>>>(xn, WkT, kraw, nullptr,
                                                                   (int)M, QD, DIMM);
    }
    {
        int tiles = (int)(M / 16) * ((VD / 16) / 3);     // v: 16x48 per wave
        gemm_bf16_kernel<3, 1><<<cdiv(tiles, 8), 256, 0, stream>>>(xn, WvT, vTbuf, bv,
                                                                   (int)M, VD, DIMM);
    }

    // 4) RoPE
    rope_q_kernel<<<cdiv((size_t)BB * SS * HH * 32, 256), 256, 0, stream>>>(qraw, qrope);
    rope_k_kernel<<<cdiv((size_t)BB * SS * 32, 256), 256, 0, stream>>>(kraw, krope);

    // 5) Attention (4096 wave-tiles, 8 waves/block)
    attn_kernel<<<512, 256, 0, stream>>>(qrope, krope, vTbuf, abias, ctx);

    // 6) Output projection + bo -> f32 out
    {
        int tiles = (int)(M / 16) * ((DIMM / 16) / 4);
        gemm_bf16_kernel<4, 2><<<cdiv(tiles, 8), 256, 0, stream>>>(ctx, WoT, out, bo,
                                                                   (int)M, DIMM, HV);
    }
}
